// MultiHeadSelfAttention_8684423872651
// MI455X (gfx1250) — matrix operand order repaired and checked by reading, not compiled, not run on hardware
//
#include <hip/hip_runtime.h>
#include <hip/hip_bf16.h>

// MHA with RoPE for MI455X (gfx1250): bf16 WMMA everywhere, f32 accumulate.
// B=4 S=2048 D=1024 H=16 DK=64. Compute-bound (~137 GFLOP vs ~9us of HBM
// traffic at 23.3 TB/s) => push all GEMMs through v_wmma_f32_16x16x32_bf16,
// and stage GEMM tiles with CDNA5 async global->LDS DMA (ASYNCcnt-tracked).

#define BB  4
#define SS  2048
#define DDIM 1024
#define HH  16
#define DKK 64

typedef __attribute__((ext_vector_type(16))) __bf16  v16bf;
typedef __attribute__((ext_vector_type(8)))  float   v8f;
typedef __attribute__((ext_vector_type(4)))  unsigned int u32x4;
typedef __attribute__((ext_vector_type(2)))  unsigned int u32x2;
typedef __attribute__((ext_vector_type(4)))  float   f32x4;

union FragBF {            // one WMMA 16-bit operand: 16 bf16 = 8 VGPRs = 32B/lane
  v16bf v;
  u32x4 q[2];
};

__device__ __forceinline__ unsigned short f32_to_bf16(float f) {
  unsigned int u = __builtin_bit_cast(unsigned int, f);
  u += 0x7fffu + ((u >> 16) & 1u);          // round-to-nearest-even
  return (unsigned short)(u >> 16);
}
__device__ __forceinline__ float bf16_to_f32(unsigned short h) {
  unsigned int u = ((unsigned int)h) << 16;
  return __builtin_bit_cast(float, u);
}

// CDNA5 async global->LDS: 16B per lane, INST_OFFSET applies to both the
// global and the LDS address. Tracked by ASYNCcnt.
__device__ __forceinline__ void async_b128(unsigned lds_off,
                                           const void* gaddr) {
  unsigned long long ga = (unsigned long long)(uintptr_t)gaddr;
  asm volatile("global_load_async_to_lds_b128 %0, %1, off"
               :: "v"(lds_off), "v"(ga) : "memory");
}
__device__ __forceinline__ void async_b128_off16(unsigned lds_off,
                                                 const void* gaddr) {
  unsigned long long ga = (unsigned long long)(uintptr_t)gaddr;
  asm volatile("global_load_async_to_lds_b128 %0, %1, off offset:16"
               :: "v"(lds_off), "v"(ga) : "memory");
}
__device__ __forceinline__ void wait_async0() {
  asm volatile("s_wait_asynccnt 0x0" ::: "memory");
}
__device__ __forceinline__ unsigned lds_off_of(const void* p) {
  // generic pointer to LDS: low 32 bits are the wave-relative LDS offset
  return (unsigned)(uintptr_t)p;
}

// ---------------------------------------------------------------- conversions
__global__ void cvt_f32_to_bf16(const float* __restrict__ src,
                                unsigned short* __restrict__ dst, int n4) {
  int i = blockIdx.x * blockDim.x + threadIdx.x;
  if (i >= n4) return;
  f32x4 f = *(const f32x4*)(src + (size_t)i * 4);
  u32x2 o;
  o.x = (unsigned)f32_to_bf16(f.x) | ((unsigned)f32_to_bf16(f.y) << 16);
  o.y = (unsigned)f32_to_bf16(f.z) | ((unsigned)f32_to_bf16(f.w) << 16);
  *(u32x2*)(dst + (size_t)i * 4) = o;
}

// ------------------------------------------------------------------ GEMM AWt
// Y[M,N] = A[M,K](bf16,row-major) * W[N,K](bf16,row-major)^T
// Block 256 threads (8 waves), tile 128x128x32. Wave = 32x64 (2x4 WMMA tiles).
// Tiles staged with global_load_async_to_lds_b128 (ASYNCcnt).
template <bool F32OUT>
__global__ __launch_bounds__(256) void gemm_xwT(
    const unsigned short* __restrict__ A, const unsigned short* __restrict__ W,
    void* __restrict__ out, int M, int N, int K) {
  constexpr int LDT = 40;  // padded LDS stride (halves): 80B -> conflict-free
  __shared__ alignas(16) unsigned short At[128 * LDT];
  __shared__ alignas(16) unsigned short Wt[128 * LDT];

  const int tid  = threadIdx.x;
  const int lane = tid & 31;
  const int wave = tid >> 5;
  const int wi   = wave >> 1;   // 0..3 along M
  const int wj   = wave & 1;    // 0..1 along N
  const int l15  = lane & 15;
  const int half = lane >> 4;
  const int m0   = blockIdx.x * 128;
  const int n0   = blockIdx.y * 128;
  const int lrow = tid >> 1;    // 0..127 (cooperative copy)
  const int lseg = tid & 1;

  const unsigned at_off =
      lds_off_of(At) + (unsigned)(lrow * LDT + lseg * 16) * 2;
  const unsigned wt_off =
      lds_off_of(Wt) + (unsigned)(lrow * LDT + lseg * 16) * 2;

  v8f acc[2][4];
#pragma unroll
  for (int i = 0; i < 2; ++i)
#pragma unroll
    for (int j = 0; j < 4; ++j) acc[i][j] = {};

  for (int k0 = 0; k0 < K; k0 += 32) {
    {  // async DMA global -> LDS, 32B per thread per tile
      const unsigned short* ga = A + (size_t)(m0 + lrow) * K + k0 + lseg * 16;
      async_b128(at_off, ga);
      async_b128_off16(at_off, ga);
      const unsigned short* gw = W + (size_t)(n0 + lrow) * K + k0 + lseg * 16;
      async_b128(wt_off, gw);
      async_b128_off16(wt_off, gw);
    }
    wait_async0();
    __syncthreads();

    FragBF af[2];
#pragma unroll
    for (int i = 0; i < 2; ++i) {  // A 16x32: lane m=l15; K segs per lane-half
      const unsigned short* base = At + (wi * 32 + i * 16 + l15) * LDT;
      af[i].q[0] = *(const u32x4*)(base + half * 16);
      af[i].q[1] = *(const u32x4*)(base + half * 16 + 8);
    }
    FragBF bf[4];
#pragma unroll
    for (int j = 0; j < 4; ++j) {  // B 32x16: lane n=l15; 16 contiguous K
      const unsigned short* base = Wt + (wj * 64 + j * 16 + l15) * LDT + half * 16;
      bf[j].q[0] = *(const u32x4*)(base);
      bf[j].q[1] = *(const u32x4*)(base + 8);
    }
#pragma unroll
    for (int i = 0; i < 2; ++i)
#pragma unroll
      for (int j = 0; j < 4; ++j)
        acc[i][j] = __builtin_amdgcn_wmma_f32_16x16x32_bf16(
            false, af[i].v, false, bf[j].v, (short)0, acc[i][j], false, false);
    __syncthreads();
  }

#pragma unroll
  for (int i = 0; i < 2; ++i)
#pragma unroll
    for (int j = 0; j < 4; ++j) {
      const int row0 = m0 + wi * 32 + i * 16 + 8 * half;  // D layout: m=r+8*half
      const int col  = n0 + wj * 64 + j * 16 + l15;
#pragma unroll
      for (int r = 0; r < 8; ++r) {
        float v = acc[i][j][r];
        if constexpr (F32OUT)
          ((float*)out)[(size_t)(row0 + r) * N + col] = v;
        else
          ((unsigned short*)out)[(size_t)(row0 + r) * N + col] = f32_to_bf16(v);
      }
    }
}

// ---------------------------------------------------------------------- RoPE
__global__ void rope_kernel(unsigned short* __restrict__ qk, int total_pairs) {
  int idx = blockIdx.x * blockDim.x + threadIdx.x;
  if (idx >= total_pairs) return;
  int row = idx / (DDIM / 2);
  int cp  = idx % (DDIM / 2);
  int i   = cp & (DKK / 2 - 1);
  int pos = row & (SS - 1);
  float inv_freq = exp2f(-(2.0f * (float)i / (float)DKK) * 13.287712379549449f);
  float ang = (float)pos * inv_freq;
  float s, c;
  __sincosf(ang, &s, &c);
  unsigned int* p = (unsigned int*)(qk + (size_t)row * DDIM + cp * 2);
  unsigned int pr = *p;
  float xe = bf16_to_f32((unsigned short)(pr & 0xffffu));
  float xo = bf16_to_f32((unsigned short)(pr >> 16));
  float re = xe * c - xo * s;
  float ro = xe * s + xo * c;
  *p = (unsigned)f32_to_bf16(re) | ((unsigned)f32_to_bf16(ro) << 16);
}

// ------------------------------------------------------------ flash attention
// Grid (B*H, S/64); 128 threads = 4 waves; wave owns 16 q rows; kv chunk = 32.
__global__ __launch_bounds__(128) void flash_attn(
    const unsigned short* __restrict__ Q, const unsigned short* __restrict__ K,
    const unsigned short* __restrict__ V, unsigned short* __restrict__ O) {
  constexpr int LDV = 40, LDP = 40;
  __shared__ alignas(16) unsigned short Vt[64 * LDV];       // [d][kv] transposed
  __shared__ alignas(16) unsigned short Pl[4 * 16 * LDP];   // per-wave P 16x32

  const int bh  = blockIdx.x;
  const int b   = bh / HH;
  const int h   = bh % HH;
  const int qb0 = blockIdx.y * 64;
  const int tid  = threadIdx.x;
  const int lane = tid & 31;
  const int wave = tid >> 5;
  const int l15  = lane & 15;
  const int half = lane >> 4;
  const size_t bs0 = (size_t)b * SS;
  const int q0 = qb0 + wave * 16;
  unsigned short* myP = Pl + wave * 16 * LDP;

  FragBF qf[2];  // Q A-frags for d=[0,32) and d=[32,64), straight from global
  {
    const unsigned short* qb = Q + (bs0 + q0 + l15) * DDIM + h * DKK;
#pragma unroll
    for (int dh = 0; dh < 2; ++dh) {
      qf[dh].q[0] = *(const u32x4*)(qb + dh * 32 + half * 16);
      qf[dh].q[1] = *(const u32x4*)(qb + dh * 32 + half * 16 + 8);
    }
  }

  v8f   oa[4];
  float mrow[8], lrow[8];
#pragma unroll
  for (int t = 0; t < 4; ++t) oa[t] = {};
#pragma unroll
  for (int r = 0; r < 8; ++r) { mrow[r] = -3.0e38f; lrow[r] = 0.0f; }

  const int nchunks = (qb0 + 64) / 32;
  const float scale = 0.125f;  // rsqrt(DK)

  for (int c = 0; c < nchunks; ++c) {
    const int kv0 = c * 32;
    __syncthreads();  // protect Vt/P reuse across iterations

    {  // stage V tile transposed: 32kv x 64d -> Vt[d][kv]
      const int kv = tid >> 2, dseg = (tid & 3) * 16;
      const unsigned short* vs = V + (bs0 + kv0 + kv) * DDIM + h * DKK + dseg;
      u32x4 v0 = *(const u32x4*)(vs);
      u32x4 v1 = *(const u32x4*)(vs + 8);
      unsigned int w[8] = {v0.x, v0.y, v0.z, v0.w, v1.x, v1.y, v1.z, v1.w};
#pragma unroll
      for (int j = 0; j < 8; ++j) {
        Vt[(dseg + 2 * j) * LDV + kv]     = (unsigned short)(w[j] & 0xffffu);
        Vt[(dseg + 2 * j + 1) * LDV + kv] = (unsigned short)(w[j] >> 16);
      }
    }

    float p[2][8], chmax[8];
#pragma unroll
    for (int r = 0; r < 8; ++r) chmax[r] = -3.0e38f;

#pragma unroll
    for (int nt = 0; nt < 2; ++nt) {  // two 16-col score tiles
      FragBF kf0, kf1;  // K B-frags (n = kv, contiguous d per lane)
      const unsigned short* kb = K + (bs0 + kv0 + nt * 16 + l15) * DDIM + h * DKK;
      kf0.q[0] = *(const u32x4*)(kb + half * 16);
      kf0.q[1] = *(const u32x4*)(kb + half * 16 + 8);
      kf1.q[0] = *(const u32x4*)(kb + 32 + half * 16);
      kf1.q[1] = *(const u32x4*)(kb + 32 + half * 16 + 8);
      v8f s = {};
      s = __builtin_amdgcn_wmma_f32_16x16x32_bf16(false, qf[0].v, false, kf0.v,
                                                  (short)0, s, false, false);
      s = __builtin_amdgcn_wmma_f32_16x16x32_bf16(false, qf[1].v, false, kf1.v,
                                                  (short)0, s, false, false);
      const int col = kv0 + nt * 16 + l15;
#pragma unroll
      for (int r = 0; r < 8; ++r) {
        const int rowq = q0 + r + 8 * half;
        float sv = s[r] * scale;
        sv = (col <= rowq) ? sv : -3.0e38f;   // causal mask
        p[nt][r] = sv;
        chmax[r] = fmaxf(chmax[r], sv);
      }
    }

#pragma unroll
    for (int r = 0; r < 8; ++r) {  // online softmax (row = 16-lane group)
      float cm = chmax[r];
#pragma unroll
      for (int m = 1; m < 16; m <<= 1) cm = fmaxf(cm, __shfl_xor(cm, m, 32));
      const float mnew  = fmaxf(mrow[r], cm);
      const float alpha = __expf(mrow[r] - mnew);
      float p0 = __expf(p[0][r] - mnew);
      float p1 = __expf(p[1][r] - mnew);
      p[0][r] = p0; p[1][r] = p1;
      float rs = p0 + p1;
#pragma unroll
      for (int m = 1; m < 16; m <<= 1) rs += __shfl_xor(rs, m, 32);
      lrow[r] = lrow[r] * alpha + rs;
      mrow[r] = mnew;
#pragma unroll
      for (int t = 0; t < 4; ++t) oa[t][r] *= alpha;
    }

#pragma unroll
    for (int nt = 0; nt < 2; ++nt)  // P: D-layout -> row-major LDS (bf16)
#pragma unroll
      for (int r = 0; r < 8; ++r)
        myP[(r + 8 * half) * LDP + nt * 16 + l15] = f32_to_bf16(p[nt][r]);

    __syncthreads();  // Vt staged + P visible

    FragBF pf;  // P as A-frag 16x32
    {
      const unsigned short* base = myP + l15 * LDP;
      pf.q[0] = *(const u32x4*)(base + half * 16);
      pf.q[1] = *(const u32x4*)(base + half * 16 + 8);
    }
#pragma unroll
    for (int t = 0; t < 4; ++t) {  // V B-frags from transposed LDS
      FragBF vf;
      const unsigned short* base = Vt + (t * 16 + l15) * LDV + half * 16;
      vf.q[0] = *(const u32x4*)(base);
      vf.q[1] = *(const u32x4*)(base + 8);
      oa[t] = __builtin_amdgcn_wmma_f32_16x16x32_bf16(
          false, pf.v, false, vf.v, (short)0, oa[t], false, false);
    }
  }

#pragma unroll
  for (int t = 0; t < 4; ++t)
#pragma unroll
    for (int r = 0; r < 8; ++r) {
      float val = oa[t][r] / lrow[r];
      O[(bs0 + q0 + r + 8 * half) * DDIM + h * DKK + t * 16 + l15] =
          f32_to_bf16(val);
    }
}

// --------------------------------------------------------------------- launch
extern "C" void kernel_launch(void* const* d_in, const int* in_sizes, int n_in,
                              void* d_out, int out_size, void* d_ws,
                              size_t ws_size, hipStream_t stream) {
  const float* x  = (const float*)d_in[0];
  const float* Wq = (const float*)d_in[1];
  const float* Wk = (const float*)d_in[2];
  const float* Wv = (const float*)d_in[3];
  const float* Wo = (const float*)d_in[4];

  const size_t xe = (size_t)BB * SS * DDIM;  // 8,388,608 elements
  const size_t we = (size_t)DDIM * DDIM;     // 1,048,576 elements

  unsigned short* xb  = (unsigned short*)d_ws;   // total ~88 MB of d_ws
  unsigned short* Wqb = xb + xe;
  unsigned short* Wkb = Wqb + we;
  unsigned short* Wvb = Wkb + we;
  unsigned short* Wob = Wvb + we;
  unsigned short* Qb  = Wob + we;
  unsigned short* Kb  = Qb + xe;
  unsigned short* Vb  = Kb + xe;
  unsigned short* Ob  = Vb + xe;

  const int xn4 = (int)(xe / 4), wn4 = (int)(we / 4);
  cvt_f32_to_bf16<<<(xn4 + 255) / 256, 256, 0, stream>>>(x, xb, xn4);
  cvt_f32_to_bf16<<<(wn4 + 255) / 256, 256, 0, stream>>>(Wq, Wqb, wn4);
  cvt_f32_to_bf16<<<(wn4 + 255) / 256, 256, 0, stream>>>(Wk, Wkb, wn4);
  cvt_f32_to_bf16<<<(wn4 + 255) / 256, 256, 0, stream>>>(Wv, Wvb, wn4);
  cvt_f32_to_bf16<<<(wn4 + 255) / 256, 256, 0, stream>>>(Wo, Wob, wn4);

  const int M = BB * SS;  // 8192
  dim3 ggrid(M / 128, DDIM / 128);
  gemm_xwT<false><<<ggrid, 256, 0, stream>>>(xb, Wqb, Qb, M, DDIM, DDIM);
  gemm_xwT<false><<<ggrid, 256, 0, stream>>>(xb, Wkb, Kb, M, DDIM, DDIM);
  gemm_xwT<false><<<ggrid, 256, 0, stream>>>(xb, Wvb, Vb, M, DDIM, DDIM);

  const int pairs = (int)(xe / 2);
  rope_kernel<<<(pairs + 255) / 256, 256, 0, stream>>>(Qb, pairs);
  rope_kernel<<<(pairs + 255) / 256, 256, 0, stream>>>(Kb, pairs);

  flash_attn<<<dim3(BB * HH, SS / 64), 128, 0, stream>>>(Qb, Kb, Vb, Ob);

  gemm_xwT<true><<<ggrid, 256, 0, stream>>>(Ob, Wob, d_out, M, DDIM, DDIM);
}
